// GNNGL_74577812128304
// MI455X (gfx1250) — compile-verified, matrix operand order checked by reading, not run on hardware
//
#include <hip/hip_runtime.h>

typedef __attribute__((ext_vector_type(16))) __bf16 v16bf;
typedef __attribute__((ext_vector_type(8)))  __bf16 v8bf;
typedef __attribute__((ext_vector_type(8)))  float  v8f;
typedef unsigned short u16;

#define NN    20000
#define LL    64
#define EE    64
#define NEDGE 640000
#define EPSF  1e-12f

__device__ __forceinline__ u16 f2bf(float x) {
    return __builtin_bit_cast(u16, (__bf16)x);
}

// ---------------------------------------------------------------------------
// Embedding stage: f[n,e] = sum_l ( w[l]*S[e][l]/||w*S[:,l]|| + G[e][l]/||G[:,l]|| )
// S[e][l] = src_emb[src[n,e]][l], G[e][l] = seg_emb[seg[n,e]][l]
// One block (64 threads) per node; 65-column LDS pad keeps both the
// column-norm pass and the row-reduce pass bank-conflict-free.
// ---------------------------------------------------------------------------
__global__ __launch_bounds__(64) void k_embed(
    const int* __restrict__ src, const int* __restrict__ seg,
    const float* __restrict__ src_emb, const float* __restrict__ seg_emb,
    const float* __restrict__ w, u16* __restrict__ f_bf)
{
    __shared__ float S[64][65];
    __shared__ float G[64][65];
    __shared__ float SC[64];
    __shared__ float GC[64];

    const int n = blockIdx.x;
    const int t = threadIdx.x;

    const int is = src[n * LL + t];
    const int ig = seg[n * LL + t];
    const float4* rs4 = (const float4*)(src_emb + (size_t)is * EE);  // 256B rows, 16B aligned
    const float4* rg4 = (const float4*)(seg_emb + (size_t)ig * EE);
    #pragma unroll
    for (int l4 = 0; l4 < EE / 4; ++l4) {
        float4 a = rs4[l4];
        float4 b = rg4[l4];
        S[t][l4 * 4 + 0] = a.x; S[t][l4 * 4 + 1] = a.y;
        S[t][l4 * 4 + 2] = a.z; S[t][l4 * 4 + 3] = a.w;
        G[t][l4 * 4 + 0] = b.x; G[t][l4 * 4 + 1] = b.y;
        G[t][l4 * 4 + 2] = b.z; G[t][l4 * 4 + 3] = b.w;
    }
    __syncthreads();

    // column t: norms over e
    const float wl = w[t];
    float ss = 0.f, sg = 0.f;
    #pragma unroll 8
    for (int e = 0; e < 64; ++e) {
        float a = wl * S[e][t];
        ss += a * a;
        float b = G[e][t];
        sg += b * b;
    }
    SC[t] = wl / fmaxf(sqrtf(ss), EPSF);
    GC[t] = 1.0f / fmaxf(sqrtf(sg), EPSF);
    __syncthreads();

    // row t: f[t] = sum_l S[t][l]*SC[l] + G[t][l]*GC[l]
    float acc = 0.f;
    #pragma unroll 8
    for (int l = 0; l < 64; ++l)
        acc += S[t][l] * SC[l] + G[t][l] * GC[l];

    f_bf[(size_t)n * 64 + t] = f2bf(acc);
}

// ---------------------------------------------------------------------------
// bf16 WMMA GEMM: C[M,N] = act(A[M,K](bf16) * B[K,N](f32->bf16) [+ bias])
// 256 threads = 8 waves; each wave owns 16 rows x N cols (N/16 accumulators),
// so A is fetched once per row-tile. B is transposed into padded LDS so each
// fragment is two 16B ds_load_b128 per lane; A fragments are two 16B
// global_load_b128 per lane. Fragment layout per ISA 7.12.2 (16-bit A 16x32):
//   lane<16 : M=lane,    K in {k0+0..7} U {k0+16..23}
//   lane>=16: M=lane-16, K in {k0+8..15} U {k0+24..31}
// B mirrors A with M<->N.
// ---------------------------------------------------------------------------
template <int K, int N, bool RELU, bool BIAS, bool OUTF32>
__global__ __launch_bounds__(256) void k_gemm(
    const u16* __restrict__ A, const float* __restrict__ B,
    const float* __restrict__ bias, void* __restrict__ Cout, int M)
{
    constexpr int KS = K + 8;                 // pad: 16B per column row-start skew
    __shared__ alignas(16) __bf16 sBT[N * KS];

    const int tid = threadIdx.x;
    #pragma unroll 4
    for (int i = tid; i < K * N; i += 256) {  // coalesced f32 read, transposed bf16 write
        const int k = i / N;
        const int c = i - k * N;
        sBT[c * KS + k] = (__bf16)B[i];
    }
    __syncthreads();

    const int wave = tid >> 5;
    const int lane = tid & 31;
    const int hi   = lane >> 4;
    const int l16  = lane & 15;

    const int rowTile = blockIdx.x * 8 + wave;
    if (rowTile * 16 >= M) return;            // wave-uniform: EXEC stays all-ones for WMMA
    const int rowBase = rowTile * 16;

    constexpr int CT = N / 16;
    v8f acc[CT];
    #pragma unroll
    for (int ct = 0; ct < CT; ++ct) acc[ct] = (v8f){};

    const v8bf* Arow = (const v8bf*)(A + (size_t)(rowBase + l16) * K);

    #pragma unroll
    for (int kk = 0; kk < K / 32; ++kk) {
        const v8bf a0 = Arow[kk * 4 + hi];      // K = kk*32 + hi*8 .. +7
        const v8bf a1 = Arow[kk * 4 + 2 + hi];  // K = kk*32 + 16 + hi*8 .. +7
        const v16bf a = __builtin_shufflevector(a0, a1,
            0, 1, 2, 3, 4, 5, 6, 7, 8, 9, 10, 11, 12, 13, 14, 15);
        #pragma unroll
        for (int ct = 0; ct < CT; ++ct) {
            const __bf16* bp = &sBT[(ct * 16 + l16) * KS + kk * 32 + hi * 8];
            const v8bf b0 = *(const v8bf*)bp;
            const v8bf b1 = *(const v8bf*)(bp + 16);
            const v16bf b = __builtin_shufflevector(b0, b1,
                0, 1, 2, 3, 4, 5, 6, 7, 8, 9, 10, 11, 12, 13, 14, 15);
            acc[ct] = __builtin_amdgcn_wmma_f32_16x16x32_bf16(
                false, a, false, b, (short)0, acc[ct], false, false);
        }
    }

    #pragma unroll
    for (int ct = 0; ct < CT; ++ct) {
        const int col = ct * 16 + l16;
        float bb = 0.0f;
        if constexpr (BIAS) bb = bias[col];
        #pragma unroll
        for (int r = 0; r < 8; ++r) {
            const int row = rowBase + r + hi * 8;  // C layout: VGPR r -> M=r (+8 for hi lanes)
            float v = acc[ct][r] + bb;
            if constexpr (RELU) v = fmaxf(v, 0.0f);
            const size_t o = (size_t)row * N + col;
            if constexpr (OUTF32) ((float*)Cout)[o] = v;
            else                  ((u16*)Cout)[o] = f2bf(v);
        }
    }
}

// ---------------------------------------------------------------------------
// Graph plumbing
// ---------------------------------------------------------------------------
__global__ void k_deg(const int* __restrict__ di, float* __restrict__ deg) {
    int i = blockIdx.x * blockDim.x + threadIdx.x;
    if (i >= NEDGE) return;
    __hip_atomic_fetch_add(&deg[di[i]], 1.0f, __ATOMIC_RELAXED, __HIP_MEMORY_SCOPE_AGENT);
}

__global__ void k_dinv(const float* __restrict__ deg, float* __restrict__ dinv) {
    int i = blockIdx.x * blockDim.x + threadIdx.x;
    if (i >= NN) return;
    dinv[i] = rsqrtf(deg[i] + 1.0f);   // +1 for self-loop
}

__global__ void k_en(const float* __restrict__ dinv, const int* __restrict__ si,
                     const int* __restrict__ di, float* __restrict__ en) {
    int i = blockIdx.x * blockDim.x + threadIdx.x;
    if (i >= NEDGE) return;
    en[i] = dinv[si[i]] * dinv[di[i]];
}

// Edge-parallel scatter-add: one thread = 4 channels of one edge.
template <int F>
__global__ void k_agg(const float* __restrict__ h, const float* __restrict__ en,
                      const int* __restrict__ si, const int* __restrict__ di,
                      float* __restrict__ agg)
{
    const int i = blockIdx.x * blockDim.x + threadIdx.x;
    constexpr int VPE = F >> 2;
    const int e = i / VPE;
    if (e >= NEDGE) return;
    const int c = (i - e * VPE) << 2;
    const int s = si[e], d = di[e];
    const float nw = en[e];
    const float4 hv = *(const float4*)(h + (size_t)s * F + c);
    float* out = agg + (size_t)d * F + c;
    __hip_atomic_fetch_add(out + 0, hv.x * nw, __ATOMIC_RELAXED, __HIP_MEMORY_SCOPE_AGENT);
    __hip_atomic_fetch_add(out + 1, hv.y * nw, __ATOMIC_RELAXED, __HIP_MEMORY_SCOPE_AGENT);
    __hip_atomic_fetch_add(out + 2, hv.z * nw, __ATOMIC_RELAXED, __HIP_MEMORY_SCOPE_AGENT);
    __hip_atomic_fetch_add(out + 3, hv.w * nw, __ATOMIC_RELAXED, __HIP_MEMORY_SCOPE_AGENT);
}

// out = relu(agg + h*dinv^2 + bias), emit bf16 for the next GEMM
template <int F>
__global__ void k_post(const float* __restrict__ h, const float* __restrict__ agg,
                       const float* __restrict__ dinv, const float* __restrict__ bias,
                       u16* __restrict__ out_bf)
{
    const int i = blockIdx.x * blockDim.x + threadIdx.x;
    if (i >= NN * F) return;
    const int n = i / F;
    const int c = i - n * F;
    const float dv = dinv[n];
    float v = agg[i] + h[i] * dv * dv + bias[c];
    v = fmaxf(v, 0.0f);
    out_bf[i] = f2bf(v);
}

// ---------------------------------------------------------------------------
extern "C" void kernel_launch(void* const* d_in, const int* in_sizes, int n_in,
                              void* d_out, int out_size, void* d_ws, size_t ws_size,
                              hipStream_t stream)
{
    const int*   src     = (const int*)d_in[0];
    const int*   seg     = (const int*)d_in[1];
    const int*   ei      = (const int*)d_in[2];
    const float* src_emb = (const float*)d_in[3];
    const float* seg_emb = (const float*)d_in[4];
    const float* w       = (const float*)d_in[5];
    const float* Wq1 = (const float*)d_in[6];  const float* bq1 = (const float*)d_in[7];
    const float* Wq2 = (const float*)d_in[8];  const float* bq2 = (const float*)d_in[9];
    const float* Wg1 = (const float*)d_in[10]; const float* bg1 = (const float*)d_in[11];
    const float* Wg2 = (const float*)d_in[12]; const float* bg2 = (const float*)d_in[13];
    const float* Wl  = (const float*)d_in[14]; const float* bl  = (const float*)d_in[15];
    const int* si = ei;
    const int* di = ei + NEDGE;

    // workspace carve-out (256B aligned)
    char* ws = (char*)d_ws;
    size_t off = 0;
    auto take = [&](size_t bytes) { char* p = ws + off; off = (off + bytes + 255) & ~(size_t)255; return p; };
    u16*   f_bf  = (u16*)  take((size_t)NN * 64 * 2);   // later reused as x3_bf (same size)
    u16*   x1_bf = (u16*)  take((size_t)NN * 128 * 2);
    u16*   q_bf  = (u16*)  take((size_t)NN * 64 * 2);
    u16*   x2_bf = (u16*)  take((size_t)NN * 128 * 2);
    float* h     = (float*)take((size_t)NN * 128 * 4);  // h1 (F=128) then h2 (F=64)
    float* agg   = (float*)take((size_t)NN * 128 * 4);  // agg1 then agg2
    float* deg   = (float*)take((size_t)NN * 4);
    float* dinv  = (float*)take((size_t)NN * 4);
    float* en    = (float*)take((size_t)NEDGE * 4);
    u16*   x3_bf = f_bf;

    const int rowTiles  = NN / 16;                  // 1250 (M divides 16 exactly)
    const int gemmBlocks = (rowTiles + 7) / 8;      // 157

    // graph normalization terms
    hipMemsetAsync(deg, 0, (size_t)NN * 4, stream);
    k_deg <<<(NEDGE + 255) / 256, 256, 0, stream>>>(di, deg);
    k_dinv<<<(NN + 255) / 256,    256, 0, stream>>>(deg, dinv);
    k_en  <<<(NEDGE + 255) / 256, 256, 0, stream>>>(dinv, si, di, en);

    // literature embedding -> f (bf16)
    k_embed<<<NN, 64, 0, stream>>>(src, seg, src_emb, seg_emb, w, f_bf);

    // query MLP
    k_gemm<64, 128, true,  true,  false><<<gemmBlocks, 256, 0, stream>>>(f_bf,  Wq1, bq1, x1_bf, NN);
    k_gemm<128, 64, false, true,  false><<<gemmBlocks, 256, 0, stream>>>(x1_bf, Wq2, bq2, q_bf,  NN);

    // GCN layer 1: h = q @ Wg1 ; aggregate ; relu(agg + h*dinv^2 + bg1)
    k_gemm<64, 128, false, false, true><<<gemmBlocks, 256, 0, stream>>>(q_bf, Wg1, nullptr, h, NN);
    hipMemsetAsync(agg, 0, (size_t)NN * 128 * 4, stream);
    k_agg<128> <<<(NEDGE * 32) / 256, 256, 0, stream>>>(h, en, si, di, agg);
    k_post<128><<<(NN * 128) / 256,   256, 0, stream>>>(h, agg, dinv, bg1, x2_bf);

    // GCN layer 2
    k_gemm<128, 64, false, false, true><<<gemmBlocks, 256, 0, stream>>>(x2_bf, Wg2, nullptr, h, NN);
    hipMemsetAsync(agg, 0, (size_t)NN * 64 * 4, stream);
    k_agg<64> <<<(NEDGE * 16) / 256, 256, 0, stream>>>(h, en, si, di, agg);
    k_post<64><<<(NN * 64) / 256,    256, 0, stream>>>(h, agg, dinv, bg2, x3_bf);

    // output linear -> f32 d_out
    k_gemm<64, 64, false, true, true><<<gemmBlocks, 256, 0, stream>>>(x3_bf, Wl, bl, (float*)d_out, NN);
}
